// ERMoEBrain_81183471829530
// MI455X (gfx1250) — matrix operand order repaired and checked by reading, not compiled, not run on hardware
//
#include <hip/hip_runtime.h>
#include <hip/hip_bf16.h>
#include <math.h>

// ---------------------------------------------------------------------------
// ERMoE brain-ViT forward for gfx1250 (MI455X).  All GEMMs use
// v_wmma_f32_16x16x32_f16 (wave32 WMMA).  Both A and B tiles are staged into
// LDS by the Tensor Data Mover (tensor_load_to_lds + s_wait_tensorcnt); all
// B operands are kept [N][K] in memory so no transpose is ever needed and all
// LDS fragment reads are contiguous 128-bit ds loads.
// ---------------------------------------------------------------------------

typedef _Float16 half_t;
typedef __attribute__((ext_vector_type(16))) _Float16     v16h;
typedef __attribute__((ext_vector_type(8)))  _Float16     v8h;
typedef __attribute__((ext_vector_type(8)))  float        v8f;
typedef __attribute__((ext_vector_type(4)))  unsigned int v4u;
typedef __attribute__((ext_vector_type(8)))  int          v8i;
typedef __attribute__((ext_vector_type(4)))  int          v4i;

constexpr int       kDim   = 256;
constexpr int       kT     = 2536;   // tokens (P+1)
constexpr int       kP     = 2535;   // patches
constexpr int       kHeads = 4;
constexpr int       kDh    = 64;
constexpr int       kFF    = 1024;
constexpr long long kVox   = 902629LL; // 91*109*91

// ----------------------------- small kernels -------------------------------

__global__ void k_cast(const float* __restrict__ s, half_t* __restrict__ d, int n) {
  int i = blockIdx.x * 256 + threadIdx.x;
  if (i < n) d[i] = (half_t)s[i];
}

// f32 [K][N] -> f16 [N][K] (weights pre-transposed once per ViT)
__global__ void k_cast_tr(const float* __restrict__ s, half_t* __restrict__ d,
                          int K, int N) {
  int i = blockIdx.x * 256 + threadIdx.x;
  if (i >= K * N) return;
  int n = i / K, k = i % K;
  d[(size_t)n * K + k] = (half_t)s[(size_t)k * N + n];
}

__global__ void k_cls(float* __restrict__ z, const float* __restrict__ cls,
                      const float* __restrict__ pos) {
  int b = blockIdx.x, c = threadIdx.x;
  z[(size_t)b * kT * kDim + c] = cls[c] + pos[c];
}

// 7x7x7 stride-7 valid conv == per-patch dot product, one channel per thread.
__global__ void k_patch_embed(const float* __restrict__ img,
                              const unsigned char* __restrict__ mask,
                              const float* __restrict__ pw,
                              const float* __restrict__ pb,
                              const float* __restrict__ pos,
                              float* __restrict__ z) {
  int p = blockIdx.x;        // 0..P-1  (d*15+h)*13+w ordering
  int b = blockIdx.y;
  int c = threadIdx.x;       // 0..255
  int pd  = p / (15 * 13);
  int rem = p % (15 * 13);
  int ph  = rem / 13;
  int pwi = rem % 13;
  const float* ib = img + (size_t)b * kVox;
  const float* wc = pw + (size_t)c * 343;
  float acc = 0.f;
  for (int dz = 0; dz < 7; ++dz)
    for (int dy = 0; dy < 7; ++dy)
      for (int dx = 0; dx < 7; ++dx) {
        size_t off = ((size_t)(pd * 7 + dz) * 109 + (ph * 7 + dy)) * 91 + (pwi * 7 + dx);
        float v = ib[off];
        if (mask) v *= (float)mask[off];
        acc += v * wc[(dz * 7 + dy) * 7 + dx];
      }
  int tkn = 1 + p;
  z[((size_t)b * kT + tkn) * kDim + c] = acc + pb[c] + pos[(size_t)tkn * kDim + c];
}

__global__ void k_layernorm(const float* __restrict__ x, const float* __restrict__ g,
                            const float* __restrict__ b, half_t* __restrict__ y16,
                            float* __restrict__ y32) {
  __shared__ float red[256];
  int row = blockIdx.x, t = threadIdx.x;
  float v = x[(size_t)row * kDim + t];
  red[t] = v; __syncthreads();
  for (int s = 128; s > 0; s >>= 1) { if (t < s) red[t] += red[t + s]; __syncthreads(); }
  float mean = red[0] * (1.f / kDim); __syncthreads();
  float d = v - mean;
  red[t] = d * d; __syncthreads();
  for (int s = 128; s > 0; s >>= 1) { if (t < s) red[t] += red[t + s]; __syncthreads(); }
  float var = red[0] * (1.f / kDim);
  float o = d * rsqrtf(var + 1e-5f) * g[t] + b[t];
  if (y16) y16[(size_t)row * kDim + t] = (half_t)o;
  if (y32) y32[(size_t)row * kDim + t] = o;
}

// qkv [B,T,768] -> Q [b,h,t,d] (x1/8), K [b,h,t,d], V transposed [b,h,d,t]
__global__ void k_qkv_split(const float* __restrict__ qkv, half_t* __restrict__ q,
                            half_t* __restrict__ k, half_t* __restrict__ vT, int total) {
  int i = blockIdx.x * 256 + threadIdx.x;
  if (i >= total) return;
  int c = i % kDim;
  int t = (i / kDim) % kT;
  int b = i / (kDim * kT);
  int h = c >> 6, d = c & 63;
  const float* src = qkv + (size_t)(b * kT + t) * (3 * kDim);
  size_t o  = ((size_t)(b * kHeads + h) * kT + t) * kDh + d;
  size_t ov = ((size_t)(b * kHeads + h) * kDh + d) * kT + t;
  q[o]   = (half_t)(src[c] * 0.125f);          // fold 1/sqrt(dh) into Q
  k[o]   = (half_t)src[kDim + c];
  vT[ov] = (half_t)src[2 * kDim + c];
}

__global__ void k_softmax(float* __restrict__ S, half_t* __restrict__ S16, int cols) {
  __shared__ float red[256];
  int row = blockIdx.x, t = threadIdx.x;
  float* r = S + (size_t)row * cols;
  float mx = -3.0e38f;
  for (int c = t; c < cols; c += 256) mx = fmaxf(mx, r[c]);
  red[t] = mx; __syncthreads();
  for (int s = 128; s > 0; s >>= 1) { if (t < s) red[t] = fmaxf(red[t], red[t + s]); __syncthreads(); }
  mx = red[0]; __syncthreads();
  float sum = 0.f;
  for (int c = t; c < cols; c += 256) { float e = __expf(r[c] - mx); r[c] = e; sum += e; }
  red[t] = sum; __syncthreads();
  for (int s = 128; s > 0; s >>= 1) { if (t < s) red[t] += red[t + s]; __syncthreads(); }
  float inv = 1.f / red[0];
  for (int c = t; c < cols; c += 256) {
    float p = r[c] * inv;
    r[c] = p;
    S16[(size_t)row * cols + c] = (half_t)p;
  }
}

// accumulate mean-over-heads attention on patch-patch block (drop CLS row/col)
__global__ void k_attn_accum(const float* __restrict__ S, float* __restrict__ Am, int first) {
  long long i = (long long)blockIdx.x * 256 + threadIdx.x;
  if (i >= (long long)kP * kP) return;
  int r = (int)(i / kP), c = (int)(i % kP);
  float v = 0.25f * S[(size_t)(r + 1) * kT + (c + 1)];
  if (first) Am[i] = v; else Am[i] += v;
}

__global__ void k_head(const float* __restrict__ hid, const float* __restrict__ hw,
                       const float* __restrict__ hb, float* __restrict__ preds, int e) {
  __shared__ float red[256];
  int b = blockIdx.x, t = threadIdx.x;
  red[t] = hid[(size_t)b * kT * kDim + t] * hw[t];  // token 0 (CLS)
  __syncthreads();
  for (int s = 128; s > 0; s >>= 1) { if (t < s) red[t] += red[t + s]; __syncthreads(); }
  if (t == 0) preds[b * 3 + e] = red[0] + hb[e];
}

// diag-energy proxy on the masked mean-attention submatrix
__global__ void k_score(const float* __restrict__ Am, const int* __restrict__ idx,
                        int n, float* __restrict__ scores, int be) {
  __shared__ float red[256];
  int t = threadIdx.x;
  float sd = 0.f;
  for (int i = t; i < n; i += 256) {
    float d = Am[(size_t)idx[i] * kP + idx[i]];
    sd += d * d;
  }
  red[t] = sd; __syncthreads();
  for (int s = 128; s > 0; s >>= 1) { if (t < s) red[t] += red[t + s]; __syncthreads(); }
  float ssd = red[0]; __syncthreads();
  float st = 0.f;
  long long nn = (long long)n * n;
  for (long long q = t; q < nn; q += 256) {
    int i = (int)(q / n), j = (int)(q % n);
    float a = Am[(size_t)idx[i] * kP + idx[j]];
    st += a * a;
  }
  red[t] = st; __syncthreads();
  for (int s = 128; s > 0; s >>= 1) { if (t < s) red[t] += red[t + s]; __syncthreads(); }
  if (t == 0) scores[be] = ssd / (red[0] + 1e-10f);
}

__global__ void k_finalize(const float* __restrict__ scores, const float* __restrict__ preds,
                           float* __restrict__ out, int Bn) {
  int b = blockIdx.x * blockDim.x + threadIdx.x;
  if (b >= Bn) return;
  float s[3];
  bool sel[3] = {false, false, false};
  for (int e = 0; e < 3; ++e) s[e] = scores[b * 3 + e];
  int ord[3] = {0, 1, 2};
  for (int i = 0; i < 2; ++i)
    for (int j = 0; j < 2 - i; ++j)
      if (s[ord[j]] < s[ord[j + 1]]) { int tmp = ord[j]; ord[j] = ord[j + 1]; ord[j + 1] = tmp; }
  int cnt = 0;
  for (int i = 0; i < 3; ++i) {
    int id = ord[i];
    if ((s[id] >= 0.5f) && (cnt < 2) && !sel[id]) { sel[id] = true; cnt++; }
  }
  if ((cnt < 2) && !sel[0]) { sel[0] = true; cnt++; }
  for (int i = 0; i < 3; ++i) {
    int id = ord[i];
    if ((cnt < 2) && !sel[id]) { sel[id] = true; cnt++; }
  }
  float ps = 0.f, wsum = 0.f;
  for (int e = 0; e < 3; ++e) {
    float w = sel[e] ? s[e] : 0.f;
    ps += preds[b * 3 + e] * w;
    wsum += w;
  }
  out[b] = ps / fmaxf(wsum, 1e-8f);
}

// ------------------------------ WMMA GEMM ----------------------------------
// C[M,N] = act(A[M,K] * B^T + bias) (+ residual); A is [M][K] row-major,
// B is [N][K] row-major (K-contiguous).  f16 in / f32 accumulate.
// Block: 256 threads (8 wave32), tile 64x64x32; each wave: one 16x32 strip.
// Both LDS tiles are 64 rows x 32 halves (8-half row pad -> 40-half stride),
// loaded by the Tensor Data Mover; fragment reads are 128-bit ds loads.

#define BM 64
#define BN 64
#define BK 32
#define LDSPITCH 40   // 32 + 8 halves; 80 B rows keep 16-B alignment

// Issue one TDM 2-D tile load: 64 rows x 32 halves, row stride `ld` elements,
// LDS rows padded by 4 DWORDs after every 16 DWORDs (-> LDSPITCH).
__device__ __forceinline__ void tdm_load_tile(const half_t* gsrc, unsigned lds_off,
                                              int rem_rows, int rem_k, int ld) {
  unsigned long long gaddr = (unsigned long long)(const void*)gsrc;
  v4u g0;
  g0[0] = 1u;                                                    // count=1
  g0[1] = lds_off;                                               // lds_addr
  g0[2] = (unsigned)gaddr;                                       // global lo
  g0[3] = (unsigned)((gaddr >> 32) & 0x01FFFFFFu) | (2u << 30);  // global hi|type=2
  unsigned td0 = (unsigned)rem_k;     // innermost extent (K)
  unsigned td1 = (unsigned)rem_rows;  // rows (M or N strip)
  v8i g1;
  // data_size=1 (2B), pad_enable, pad_interval=3 (16 DW), pad_amount=3 (4 DW)
  g1[0] = (int)((1u << 16) | (1u << 20) | (3u << 22) | (3u << 25));
  g1[1] = (int)((td0 & 0xFFFFu) << 16);                 // tensor_dim0 lo
  g1[2] = (int)((td0 >> 16) | ((td1 & 0xFFFFu) << 16)); // dim0 hi | dim1 lo
  g1[3] = (int)((td1 >> 16) | ((unsigned)BK << 16));    // dim1 hi | tile_dim0=32
  g1[4] = (int)(unsigned)BM;                            // tile_dim1=64, tile_dim2=0
  g1[5] = (int)(unsigned)ld;                            // tensor_dim0_stride lo
  g1[6] = 0;
  g1[7] = 0;
  v4i g2 = {0, 0, 0, 0};
  v4i g3 = {0, 0, 0, 0};
#if defined(__clang_major__) && (__clang_major__ >= 23)
  v8i g4 = {0, 0, 0, 0, 0, 0, 0, 0};
  __builtin_amdgcn_tensor_load_to_lds(g0, g1, g2, g3, g4, 0);
#else
  __builtin_amdgcn_tensor_load_to_lds(g0, g1, g2, g3, 0);
#endif
}

__global__ __launch_bounds__(256) void k_gemm(
    const half_t* __restrict__ A, const half_t* __restrict__ Bm,
    const float* __restrict__ bias, const float* __restrict__ residual,
    float* __restrict__ C32, half_t* __restrict__ C16,
    int M, int N, int K, int lda, int ldb, int ldc, int act) {
  __shared__ __align__(16) half_t As[BM][LDSPITCH];   // [m][k]
  __shared__ __align__(16) half_t BsT[BN][LDSPITCH];  // [n][k]

  const int tid  = threadIdx.x;
  const int lane = tid & 31;
  const int wid  = tid >> 5;
  const int wm   = wid & 3;    // 16-row strip within block tile
  const int wn   = wid >> 2;   // 32-col strip within block tile
  const int m0   = blockIdx.y * BM;
  const int n0   = blockIdx.x * BN;

  v8f acc[2];
#pragma unroll
  for (int t2 = 0; t2 < 2; ++t2)
#pragma unroll
    for (int i = 0; i < 8; ++i) acc[t2][i] = 0.f;

  const int mi = lane & 15;
  const int g  = lane >> 4;

  for (int k0 = 0; k0 < K; k0 += BK) {
    if (tid == 0) {
      tdm_load_tile(A  + (size_t)m0 * lda + k0,
                    (unsigned)(unsigned long long)(void*)&As[0][0],
                    M - m0, K - k0, lda);
      tdm_load_tile(Bm + (size_t)n0 * ldb + k0,
                    (unsigned)(unsigned long long)(void*)&BsT[0][0],
                    N - n0, K - k0, ldb);
    }
    if ((k0 + BK) < K) {           // speculative prefetch of next K-step rows
      if (tid < BM)
        __builtin_prefetch(A + (size_t)(m0 + tid) * lda + (k0 + BK), 0, 1);
      else if (tid < BM + BN)
        __builtin_prefetch(Bm + (size_t)(n0 + tid - BM) * ldb + (k0 + BK), 0, 1);
    }
    if (tid == 0) __builtin_amdgcn_s_wait_tensorcnt(0);
    __syncthreads();

    // ---- fragments per ISA 7.12.2 wave32 layouts: contiguous 128-bit loads ----
    const half_t* ap = &As[wm * 16 + mi][0];
    v8h a0 = *(const v8h*)(ap + g * 8);
    v8h a1 = *(const v8h*)(ap + 16 + g * 8);
    v16h af = __builtin_shufflevector(a0, a1, 0, 1, 2, 3, 4, 5, 6, 7,
                                      8, 9, 10, 11, 12, 13, 14, 15);
#pragma unroll
    for (int t2 = 0; t2 < 2; ++t2) {
      const half_t* bp = &BsT[wn * 32 + t2 * 16 + mi][g * 8];
      v8h b0 = *(const v8h*)bp;
      v8h b1 = *(const v8h*)(bp + 16);
      v16h bf = __builtin_shufflevector(b0, b1, 0, 1, 2, 3, 4, 5, 6, 7,
                                        8, 9, 10, 11, 12, 13, 14, 15);
      acc[t2] = __builtin_amdgcn_wmma_f32_16x16x32_f16(
          false, af, false, bf, (short)0, acc[t2], false, false);
    }
    __syncthreads();
  }

  // ---- epilogue: bias, GELU, residual, f32/f16 stores ----
#pragma unroll
  for (int t2 = 0; t2 < 2; ++t2) {
    int col = n0 + wn * 32 + t2 * 16 + mi;
    if (col >= N) continue;
#pragma unroll
    for (int v = 0; v < 8; ++v) {
      int row = m0 + wm * 16 + g * 8 + v;
      if (row >= M) continue;
      float val = acc[t2][v];
      if (bias) val += bias[col];
      if (act) val = 0.5f * val * (1.f + erff(val * 0.70710678118f));
      if (residual) val += residual[(size_t)row * ldc + col];
      if (C32) C32[(size_t)row * ldc + col] = val;
      if (C16) C16[(size_t)row * ldc + col] = (half_t)val;
    }
  }
}

// ------------------------------- host side ---------------------------------

struct LayerP {
  const float *ln1_g, *ln1_b, *qkv_w, *qkv_b, *proj_w, *proj_b;
  const float *ln2_g, *ln2_b, *fc1_w, *fc1_b, *fc2_w, *fc2_b;
};
struct ViTP {
  const float *patch_w, *patch_b, *cls, *pos, *lnf_g, *lnf_b;
  LayerP L[4];
};

struct Cursor {
  void* const* leaves; int li;
  const float* base; size_t off;
  bool packed;
  const float* take(size_t n) {
    if (packed) { const float* p = base + off; off += n; return p; }
    return (const float*)leaves[li++];
  }
};

// alphabetical (jax tree_leaves) leaf order
static void parse_vit(Cursor& c, ViTP& v) {
  v.cls = c.take(256);
  for (int l = 0; l < 4; ++l) {
    LayerP& L = v.L[l];
    L.fc1_b = c.take(1024);  L.fc1_w = c.take(262144);
    L.fc2_b = c.take(256);   L.fc2_w = c.take(262144);
    L.ln1_b = c.take(256);   L.ln1_g = c.take(256);
    L.ln2_b = c.take(256);   L.ln2_g = c.take(256);
    L.proj_b = c.take(256);  L.proj_w = c.take(65536);
    L.qkv_b = c.take(768);   L.qkv_w = c.take(196608);
  }
  v.lnf_b = c.take(256);  v.lnf_g = c.take(256);
  v.patch_b = c.take(256); v.patch_w = c.take(87808);
  v.pos = c.take(649216);
}

extern "C" void kernel_launch(void* const* d_in, const int* in_sizes, int n_in,
                              void* d_out, int out_size, void* d_ws, size_t ws_size,
                              hipStream_t stream) {
  (void)out_size; (void)ws_size;

  const float* images = (const float*)d_in[0];
  ViTP sharedV, experts[3];
  const float *head_w, *head_b;
  const int* idx[3]; int nidx[3];
  const unsigned char* rvox;

  if (n_in >= 223) {  // one pointer per pytree leaf
    Cursor c{d_in, 1, nullptr, 0, false};
    parse_vit(c, sharedV);
    for (int e = 0; e < 3; ++e) parse_vit(c, experts[e]);
    head_w = (const float*)d_in[c.li]; c.li++;
    head_b = (const float*)d_in[c.li]; c.li++;
    for (int e = 0; e < 3; ++e) { idx[e] = (const int*)d_in[c.li]; nidx[e] = in_sizes[c.li]; c.li++; }
    rvox = (const unsigned char*)d_in[c.li];
  } else {            // one flat buffer per top-level dict entry
    Cursor cs{nullptr, 0, (const float*)d_in[1], 0, true};
    parse_vit(cs, sharedV);
    Cursor ce{nullptr, 0, (const float*)d_in[2], 0, true};
    for (int e = 0; e < 3; ++e) parse_vit(ce, experts[e]);
    head_w = (const float*)d_in[3];
    head_b = (const float*)d_in[4];
    const int* all = (const int*)d_in[5];
    int tot = in_sizes[5];
    int third = tot / 3;
    nidx[0] = tot - 2 * third; nidx[1] = third; nidx[2] = third;
    idx[0] = all; idx[1] = all + nidx[0]; idx[2] = all + nidx[0] + nidx[1];
    rvox = (const unsigned char*)d_in[6];
  }

  int Bn = (int)((long long)in_sizes[0] / kVox);
  if (Bn <= 0) Bn = 2;

  // ---- workspace bump allocation (~145 MB) ----
  char* wp = (char*)d_ws;
  auto bump = [&](size_t bytes) -> void* {
    void* r = wp; wp += (bytes + 255) & ~(size_t)255; return r;
  };
  half_t* wq16 = (half_t*)bump((size_t)4 * 196608 * 2);  // [768][256] per layer
  half_t* wp16 = (half_t*)bump((size_t)4 * 65536 * 2);   // [256][256]
  half_t* w116 = (half_t*)bump((size_t)4 * 262144 * 2);  // [1024][256]
  half_t* w216 = (half_t*)bump((size_t)4 * 262144 * 2);  // [256][1024]
  float*  z     = (float*) bump((size_t)Bn * kT * kDim * 4);
  half_t* h16   = (half_t*)bump((size_t)Bn * kT * kDim * 2);
  float*  qkv32 = (float*) bump((size_t)Bn * kT * 3 * kDim * 4);
  half_t* q16   = (half_t*)bump((size_t)Bn * kT * kDim * 2);
  half_t* k16   = (half_t*)bump((size_t)Bn * kT * kDim * 2);
  half_t* vT16  = (half_t*)bump((size_t)Bn * kT * kDim * 2);  // [b,h,d,t]
  float*  S32   = (float*) bump((size_t)kT * kT * 4);
  half_t* S16   = (half_t*)bump((size_t)kT * kT * 2);
  half_t* o16   = (half_t*)bump((size_t)Bn * kT * kDim * 2);
  half_t* g16   = (half_t*)bump((size_t)Bn * kT * kFF * 2);
  float*  hid32 = (float*) bump((size_t)Bn * kT * kDim * 4);
  float*  attnm = (float*) bump((size_t)Bn * kP * kP * 4);
  float*  preds = (float*) bump(64);
  float*  scor  = (float*) bump(64);

  auto gemm = [&](const half_t* A, const half_t* Bm, const float* bias,
                  const float* residual, float* C32, half_t* C16,
                  int M, int N, int K, int lda, int ldb, int ldc, int act) {
    dim3 grid((N + BN - 1) / BN, (M + BM - 1) / BM);
    k_gemm<<<grid, 256, 0, stream>>>(A, Bm, bias, residual, C32, C16,
                                     M, N, K, lda, ldb, ldc, act);
  };
  auto cast_tr = [&](const float* s, half_t* d, int K, int N) {
    k_cast_tr<<<(K * N + 255) / 256, 256, 0, stream>>>(s, d, K, N);
  };

  auto run_vit = [&](const ViTP& V, const unsigned char* mask, bool want_attn) {
    for (int l = 0; l < 4; ++l) {
      cast_tr(V.L[l].qkv_w,  wq16 + (size_t)l * 196608, kDim, 3 * kDim);
      cast_tr(V.L[l].proj_w, wp16 + (size_t)l * 65536,  kDim, kDim);
      cast_tr(V.L[l].fc1_w,  w116 + (size_t)l * 262144, kDim, kFF);
      cast_tr(V.L[l].fc2_w,  w216 + (size_t)l * 262144, kFF, kDim);
    }
    k_cls<<<Bn, 256, 0, stream>>>(z, V.cls, V.pos);
    k_patch_embed<<<dim3(kP, Bn), 256, 0, stream>>>(images, mask, V.patch_w,
                                                    V.patch_b, V.pos, z);
    int M = Bn * kT;
    for (int l = 0; l < 4; ++l) {
      const LayerP& L = V.L[l];
      k_layernorm<<<M, 256, 0, stream>>>(z, L.ln1_g, L.ln1_b, h16, nullptr);
      gemm(h16, wq16 + (size_t)l * 196608, L.qkv_b, nullptr, qkv32, nullptr,
           M, 3 * kDim, kDim, kDim, kDim, 3 * kDim, 0);
      k_qkv_split<<<(M * kDim + 255) / 256, 256, 0, stream>>>(qkv32, q16, k16, vT16, M * kDim);
      for (int b = 0; b < Bn; ++b)
        for (int h = 0; h < kHeads; ++h) {
          size_t hoff = (size_t)(b * kHeads + h) * kT * kDh;
          gemm(q16 + hoff, k16 + hoff, nullptr, nullptr, S32, nullptr,
               kT, kT, kDh, kDh, kDh, kT, 0);
          k_softmax<<<kT, 256, 0, stream>>>(S32, S16, kT);
          if (want_attn && l == 3)
            k_attn_accum<<<(int)(((long long)kP * kP + 255) / 256), 256, 0, stream>>>(
                S32, attnm + (size_t)b * kP * kP, h == 0 ? 1 : 0);
          gemm(S16, vT16 + hoff, nullptr, nullptr, nullptr,
               o16 + (size_t)b * kT * kDim + h * kDh,
               kT, kDh, kT, kT, kT, kDim, 0);
        }
      gemm(o16, wp16 + (size_t)l * 65536, L.proj_b, z, z, nullptr,
           M, kDim, kDim, kDim, kDim, kDim, 0);
      k_layernorm<<<M, 256, 0, stream>>>(z, L.ln2_g, L.ln2_b, h16, nullptr);
      gemm(h16, w116 + (size_t)l * 262144, L.fc1_b, nullptr, nullptr, g16,
           M, kFF, kDim, kDim, kDim, kFF, 1);
      gemm(g16, w216 + (size_t)l * 262144, L.fc2_b, z, z, nullptr,
           M, kDim, kFF, kFF, kFF, kDim, 0);
    }
    k_layernorm<<<M, 256, 0, stream>>>(z, V.lnf_g, V.lnf_b, nullptr, hid32);
  };

  // shared ViT (produces attnm), then 3 masked experts (produce preds)
  run_vit(sharedV, nullptr, true);
  for (int e = 0; e < 3; ++e) {
    run_vit(experts[e], rvox + (size_t)e * kVox, false);
    k_head<<<Bn, 256, 0, stream>>>(hid32, head_w + (size_t)e * kDim, head_b, preds, e);
  }

  for (int e = 0; e < 3; ++e)
    for (int b = 0; b < Bn; ++b)
      k_score<<<1, 256, 0, stream>>>(attnm + (size_t)b * kP * kP,
                                     idx[e], nidx[e], scor, b * 3 + e);

  k_finalize<<<1, 64, 0, stream>>>(scor, preds, (float*)d_out, Bn);
}
